// MultiHeadLatentAttention_7559142441607
// MI455X (gfx1250) — matrix operand order repaired and checked by reading, not compiled, not run on hardware
//
#include <hip/hip_runtime.h>
#include <hip/hip_bf16.h>
#include <cstdint>
#include <cstddef>

// ---------------------------------------------------------------------------
// MLA forward for MI455X (gfx1250), wave32, WMMA bf16 path.
// GEMMs: C(f32) = A(bf16, MxK row-major) @ Bt(bf16, NxK row-major, weights
// pre-transposed) via v_wmma_f32_16x16x32_bf16 with double-buffered
// global_load_async_to_lds_b128 tile staging (ASYNCcnt).
// Attention = flash style, online softmax, direct-to-VGPR fragments.
// ---------------------------------------------------------------------------

typedef __attribute__((ext_vector_type(16))) __bf16       v16bf;
typedef __attribute__((ext_vector_type(8)))  float        v8f;
typedef __attribute__((ext_vector_type(4)))  unsigned int u32x4;

union BFrag { v16bf v; u32x4 q[2]; };

static constexpr int B_   = 2;
static constexpr int T_   = 2048;
static constexpr int D_   = 2048;
static constexpr int H_   = 16;
static constexpr int ND_  = 128;
static constexpr int RD_  = 64;
static constexpr int VD_  = 128;
static constexpr int QR_  = 1536;
static constexpr int KVR_ = 512;
static constexpr int BT_  = B_ * T_;
static constexpr int DQK_ = ND_ + RD_;   // 192

// ---------------------------------------------------------------------------
// f32 -> bf16 conversion (plain, for activations x)
// ---------------------------------------------------------------------------
__global__ void cvt_f32_bf16_kernel(const float* __restrict__ in,
                                    __bf16* __restrict__ out, size_t n) {
  size_t i = (size_t)blockIdx.x * blockDim.x + threadIdx.x;
  size_t stride = (size_t)gridDim.x * blockDim.x;
  for (; i < n; i += stride) out[i] = (__bf16)in[i];
}

// ---------------------------------------------------------------------------
// f32 -> bf16 transpose-convert: in (R x Ccol) f32 -> out (Ccol x R) bf16.
// 32x32 LDS tile, 256 threads. R, Ccol multiples of 32 (true for all weights).
// ---------------------------------------------------------------------------
__global__ __launch_bounds__(256) void cvt_transpose_kernel(
    const float* __restrict__ in, __bf16* __restrict__ out, int R, int Ccol) {
  __shared__ __bf16 tile[32][33];
  const int tx = threadIdx.x & 31;
  const int ty = threadIdx.x >> 5;           // 0..7
  const int rb = blockIdx.y * 32;
  const int cb = blockIdx.x * 32;
#pragma unroll
  for (int i = 0; i < 32; i += 8)
    tile[ty + i][tx] = (__bf16)in[(size_t)(rb + ty + i) * Ccol + cb + tx];
  __syncthreads();
#pragma unroll
  for (int i = 0; i < 32; i += 8)
    out[(size_t)(cb + ty + i) * R + rb + tx] = tile[tx][ty + i];
}

// ---------------------------------------------------------------------------
// Async global -> LDS 16B copy (CDNA5 TDM-family path, tracked by ASYNCcnt).
// lds_off: 32-bit LDS byte offset; gaddr: 64-bit global address.
// ---------------------------------------------------------------------------
__device__ __forceinline__ void async_copy_b128(unsigned lds_off,
                                                unsigned long long gaddr) {
  asm volatile("global_load_async_to_lds_b128 %0, %1, off"
               :
               : "v"(lds_off), "v"(gaddr)
               : "memory");
}
__device__ __forceinline__ void wait_asynccnt0() {
  asm volatile("s_wait_asynccnt 0" ::: "memory");
}

// ---------------------------------------------------------------------------
// bf16 GEMM: block tile 128(M) x 128(N), K-tile 64 (two 32-k WMMA steps),
// 256 threads = 8 waves in a 4(M) x 2(N) wave grid (each wave 32x64 = 2x4
// fragments). A: MxK row-major. Bt: NxK row-major (weights pre-transposed).
// Double-buffered async tile staging. M % 128 == 0, K % 64 == 0 (true for all
// launches); N guarded (needed for N=64).
// ---------------------------------------------------------------------------
#define LP 72   // padded LDS row stride in elements (144B: 16B-aligned, no bank conflicts)

__global__ __launch_bounds__(256) void gemm_bf16_kernel(
    const __bf16* __restrict__ A, const __bf16* __restrict__ Bt,
    float* __restrict__ C, int M, int N, int K) {
  __shared__ __bf16 lds_a[2][128][LP];
  __shared__ __bf16 lds_b[2][128][LP];

  const int tid  = threadIdx.x;
  const int wave = tid >> 5;
  const int lane = tid & 31;
  const int hl   = lane >> 4;
  const int l15  = lane & 15;
  const int bm   = blockIdx.y * 128;
  const int bn   = blockIdx.x * 128;
  const int wm   = (wave >> 1) * 32;      // 0,32,64,96
  const int wn   = (wave & 1) * 64;       // 0,64

  v8f acc[2][4] = {};

  // Issue one 128x64 tile of A and Bt into LDS buffer p (async).
  auto issue_tile = [&](int k0, int p) {
#pragma unroll
    for (int it = 0; it < 4; ++it) {
      const int i = tid + it * 256;       // 0..1023
      const int r = i >> 3;               // 0..127
      const int c = (i & 7) * 8;          // 0..56
      const unsigned loff = (unsigned)((r * LP + c) * 2);
      async_copy_b128(
          (unsigned)(size_t)&lds_a[p][0][0] + loff,
          (unsigned long long)(size_t)(A + (size_t)(bm + r) * K + k0 + c));
      if (bn + r < N)
        async_copy_b128(
            (unsigned)(size_t)&lds_b[p][0][0] + loff,
            (unsigned long long)(size_t)(Bt + (size_t)(bn + r) * K + k0 + c));
    }
  };

  issue_tile(0, 0);
  int p = 0;
  for (int k0 = 0; k0 < K; k0 += 64) {
    wait_asynccnt0();       // tile k0 landed (this wave's copies)
    __syncthreads();        // all waves' copies landed; prev compute finished
    if (k0 + 64 < K) issue_tile(k0 + 64, p ^ 1);

#pragma unroll
    for (int s = 0; s < 2; ++s) {         // two 32-k WMMA steps
      v16bf af[2];
#pragma unroll
      for (int i = 0; i < 2; ++i) {
        const int row = wm + i * 16 + l15;
        BFrag f;
        f.q[0] = *(const u32x4*)(&lds_a[p][row][s * 32 + hl * 8]);
        f.q[1] = *(const u32x4*)(&lds_a[p][row][s * 32 + hl * 8 + 16]);
        af[i] = f.v;
      }
      v16bf bfr[4];
#pragma unroll
      for (int j = 0; j < 4; ++j) {
        const int nn = wn + j * 16 + l15;
        BFrag f;
        f.q[0] = *(const u32x4*)(&lds_b[p][nn][s * 32 + hl * 8]);
        f.q[1] = *(const u32x4*)(&lds_b[p][nn][s * 32 + hl * 8 + 16]);
        bfr[j] = f.v;
      }
#pragma unroll
      for (int i = 0; i < 2; ++i)
#pragma unroll
        for (int j = 0; j < 4; ++j)
          acc[i][j] = __builtin_amdgcn_wmma_f32_16x16x32_bf16(
              false, af[i], false, bfr[j], (short)0, acc[i][j], false, false);
    }
    p ^= 1;
  }

  // C layout: VGPR r -> row r + hl*8; lane l15 -> column.
#pragma unroll
  for (int i = 0; i < 2; ++i)
#pragma unroll
    for (int j = 0; j < 4; ++j)
#pragma unroll
      for (int r = 0; r < 8; ++r) {
        const int row = bm + wm + i * 16 + hl * 8 + r;
        const int col = bn + wn + j * 16 + l15;
        if (col < N) C[(size_t)row * N + col] = acc[i][j][r];
      }
}

// ---------------------------------------------------------------------------
// Row-wise RMSNorm -> bf16.  One block (256 threads) per row.
// ---------------------------------------------------------------------------
__global__ __launch_bounds__(256) void rmsnorm_kernel(
    const float* __restrict__ in, const float* __restrict__ g,
    __bf16* __restrict__ out, int N) {
  const int row = blockIdx.x;
  const float* r = in + (size_t)row * N;
  float ss = 0.0f;
  for (int i = threadIdx.x; i < N; i += 256) { float v = r[i]; ss += v * v; }
#pragma unroll
  for (int m = 16; m; m >>= 1) ss += __shfl_xor(ss, m, 32);
  __shared__ float part[8];
  if ((threadIdx.x & 31) == 0) part[threadIdx.x >> 5] = ss;
  __syncthreads();
  float tot = 0.0f;
#pragma unroll
  for (int w = 0; w < 8; ++w) tot += part[w];
  const float inv = rsqrtf(tot / (float)N + 1e-6f);
  for (int i = threadIdx.x; i < N; i += 256)
    out[(size_t)row * N + i] = (__bf16)(r[i] * inv * g[i]);
}

// ---------------------------------------------------------------------------
// RoPE on k_rope (B*T rows x 64). One block (32 threads = pairs) per row.
// ---------------------------------------------------------------------------
__global__ void rope_k_kernel(const float* __restrict__ in,
                              float* __restrict__ out) {
  const int row = blockIdx.x;           // b*T + t
  const int t   = row % T_;
  const int i   = threadIdx.x;          // pair 0..31
  const float x1 = in[(size_t)row * RD_ + 2 * i];
  const float x2 = in[(size_t)row * RD_ + 2 * i + 1];
  const float invf = __expf(-((float)(2 * i) / (float)RD_) * 9.210340372f);
  const float a = (float)t * invf;
  const float c = __cosf(a), s = __sinf(a);
  out[(size_t)row * RD_ + 2 * i]     = x1 * c - x2 * s;
  out[(size_t)row * RD_ + 2 * i + 1] = x1 * s + x2 * c;
}

// ---------------------------------------------------------------------------
// RoPE q_rope + pack q -> bf16 (B,H,T,192). One block (192 thr) per (b,h,t).
// ---------------------------------------------------------------------------
__global__ void rope_pack_q_kernel(const float* __restrict__ qn,   // (BT, H*128)
                                   const float* __restrict__ qr,   // (BT, H*64)
                                   __bf16* __restrict__ q) {       // (B,H,T,192)
  const int idx = blockIdx.x;                 // b*H*T + h*T + t
  const int b = idx / (H_ * T_);
  const int h = (idx / T_) % H_;
  const int t = idx % T_;
  const size_t row = (size_t)b * T_ + t;
  const int j = threadIdx.x;                  // 0..191
  float val;
  if (j < ND_) {
    val = qn[row * (H_ * ND_) + h * ND_ + j];
  } else {
    const int jr = j - ND_;
    const int i  = jr >> 1;
    const float x1 = qr[row * (H_ * RD_) + h * RD_ + 2 * i];
    const float x2 = qr[row * (H_ * RD_) + h * RD_ + 2 * i + 1];
    const float invf = __expf(-((float)(2 * i) / (float)RD_) * 9.210340372f);
    const float a = (float)t * invf;
    const float c = __cosf(a), s = __sinf(a);
    val = (jr & 1) ? (x1 * s + x2 * c) : (x1 * c - x2 * s);
  }
  q[((size_t)(b * H_ + h) * T_ + t) * DQK_ + j] = (__bf16)val;
}

// ---------------------------------------------------------------------------
// Pack k (nope + broadcast roped k_rope) and v:
//  - f32 k output (B,H,T,192), f32 v_t output (B,H,T,128)  [parts of d_out]
//  - bf16 k (B,H,T,192) and transposed bf16 v (B,H,128,T) for attention
// ---------------------------------------------------------------------------
__global__ void pack_kv_kernel(const float* __restrict__ kn,  // (BT, H*128)
                               const float* __restrict__ kr,  // (BT, 64) roped
                               const float* __restrict__ vr,  // (BT, H*128)
                               float* __restrict__ k_out,     // (B,H,T,192)
                               float* __restrict__ v_out,     // (B,H,T,128)
                               __bf16* __restrict__ k_bf,     // (B,H,T,192)
                               __bf16* __restrict__ v_bt) {   // (B,H,128,T)
  const int idx = blockIdx.x;
  const int b = idx / (H_ * T_);
  const int h = (idx / T_) % H_;
  const int t = idx % T_;
  const size_t row = (size_t)b * T_ + t;
  const size_t bht = (size_t)(b * H_ + h) * T_ + t;
  const int j = threadIdx.x;
  const float kv = (j < ND_) ? kn[row * (H_ * ND_) + h * ND_ + j]
                             : kr[row * RD_ + (j - ND_)];
  k_out[bht * DQK_ + j] = kv;
  k_bf [bht * DQK_ + j] = (__bf16)kv;
  if (j < VD_) {
    const float vv = vr[row * (H_ * VD_) + h * VD_ + j];
    v_out[bht * VD_ + j] = vv;
    v_bt[((size_t)(b * H_ + h) * VD_ + j) * T_ + t] = (__bf16)vv;
  }
}

// ---------------------------------------------------------------------------
// Flash attention. Grid (T/128, H, B), 256 threads = 8 waves.
// Each wave owns 16 query rows; loops over 64-key tiles with online softmax.
// Q/K: (B,H,T,192) bf16; Vt: (B,H,128,T) bf16; out (B,T,H*128) bf16.
// ---------------------------------------------------------------------------
__global__ __launch_bounds__(256) void flash_attn_kernel(
    const __bf16* __restrict__ q, const __bf16* __restrict__ k,
    const __bf16* __restrict__ vt, __bf16* __restrict__ o) {
  __shared__ __bf16 pbuf[8][16][64];   // per-wave P^T staging (16KB)

  const int b = blockIdx.z, h = blockIdx.y;
  const int wave = threadIdx.x >> 5;
  const int lane = threadIdx.x & 31;
  const int hl = lane >> 4, l15 = lane & 15;
  const int qbase = blockIdx.x * 128 + wave * 16;
  const size_t bh = (size_t)(b * H_ + h);
  const __bf16* kbase = k + bh * T_ * DQK_;
  const __bf16* vbase = vt + bh * VD_ * T_;
  const float scale = 0.07216878365f;   // 1/sqrt(192)

  v16bf qf[6];
  {
    const __bf16* p = q + (bh * T_ + qbase + l15) * DQK_ + hl * 8;
#pragma unroll
    for (int c = 0; c < 6; ++c) {
      BFrag f;
      f.q[0] = *(const u32x4*)(p + c * 32);
      f.q[1] = *(const u32x4*)(p + c * 32 + 16);
      qf[c] = f.v;
    }
  }

  v8f oacc[8] = {};
  float mrow[8], lrow[8];
#pragma unroll
  for (int r = 0; r < 8; ++r) { mrow[r] = -3.0e38f; lrow[r] = 0.0f; }

  const int qhi = qbase + 15;
  for (int t0 = 0; t0 <= qhi; t0 += 64) {
    v8f sacc[4] = {};
#pragma unroll
    for (int c = 0; c < 6; ++c) {
#pragma unroll
      for (int j = 0; j < 4; ++j) {
        const __bf16* kp =
            kbase + (size_t)(t0 + j * 16 + l15) * DQK_ + hl * 8 + c * 32;
        BFrag f;
        f.q[0] = *(const u32x4*)kp;
        f.q[1] = *(const u32x4*)(kp + 16);
        sacc[j] = __builtin_amdgcn_wmma_f32_16x16x32_bf16(
            false, qf[c], false, f.v, (short)0, sacc[j], false, false);
      }
    }

    float pv[4][8];
    float rmax[8];
#pragma unroll
    for (int r = 0; r < 8; ++r) rmax[r] = -3.0e38f;
#pragma unroll
    for (int j = 0; j < 4; ++j)
#pragma unroll
      for (int r = 0; r < 8; ++r) {
        const int tcol = t0 + j * 16 + l15;
        const int qrow = qbase + hl * 8 + r;
        const float s = (tcol <= qrow) ? sacc[j][r] * scale : -3.0e38f;
        pv[j][r] = s;
        rmax[r] = fmaxf(rmax[r], s);
      }
#pragma unroll
    for (int r = 0; r < 8; ++r) {
#pragma unroll
      for (int m = 1; m < 16; m <<= 1)
        rmax[r] = fmaxf(rmax[r], __shfl_xor(rmax[r], m, 32));
    }
    float alpha[8];
#pragma unroll
    for (int r = 0; r < 8; ++r) {
      const float mn = fmaxf(mrow[r], rmax[r]);
      alpha[r] = __expf(mrow[r] - mn);
      mrow[r] = mn;
    }
#pragma unroll
    for (int j = 0; j < 4; ++j)
#pragma unroll
      for (int r = 0; r < 8; ++r) pv[j][r] = __expf(pv[j][r] - mrow[r]);
#pragma unroll
    for (int r = 0; r < 8; ++r) {
      float s = pv[0][r] + pv[1][r] + pv[2][r] + pv[3][r];
#pragma unroll
      for (int m = 1; m < 16; m <<= 1) s += __shfl_xor(s, m, 32);
      lrow[r] = lrow[r] * alpha[r] + s;
    }
#pragma unroll
    for (int n = 0; n < 8; ++n)
#pragma unroll
      for (int r = 0; r < 8; ++r) oacc[n][r] *= alpha[r];

    // transpose P (C layout -> A layout) through per-wave LDS
    asm volatile("" ::: "memory");
#pragma unroll
    for (int j = 0; j < 4; ++j)
#pragma unroll
      for (int r = 0; r < 8; ++r)
        pbuf[wave][hl * 8 + r][j * 16 + l15] = (__bf16)pv[j][r];
    asm volatile("s_wait_dscnt 0" ::: "memory");

    // O += P @ V  (K=64 => two 32-wide WMMA steps)
#pragma unroll
    for (int kk = 0; kk < 2; ++kk) {
      BFrag pa;
      const __bf16* pp = &pbuf[wave][l15][kk * 32 + hl * 8];
      pa.q[0] = *(const u32x4*)pp;
      pa.q[1] = *(const u32x4*)(pp + 16);
#pragma unroll
      for (int n = 0; n < 8; ++n) {
        const __bf16* vp =
            vbase + (size_t)(n * 16 + l15) * T_ + t0 + kk * 32 + hl * 8;
        BFrag vb;
        vb.q[0] = *(const u32x4*)vp;
        vb.q[1] = *(const u32x4*)(vp + 16);
        oacc[n] = __builtin_amdgcn_wmma_f32_16x16x32_bf16(
            false, pa.v, false, vb.v, (short)0, oacc[n], false, false);
      }
    }
  }

  float invl[8];
#pragma unroll
  for (int r = 0; r < 8; ++r) invl[r] = 1.0f / lrow[r];
#pragma unroll
  for (int n = 0; n < 8; ++n)
#pragma unroll
    for (int r = 0; r < 8; ++r) {
      const int trow = qbase + hl * 8 + r;
      const int d = n * 16 + l15;
      o[((size_t)b * T_ + trow) * (H_ * VD_) + h * VD_ + d] =
          (__bf16)(oacc[n][r] * invl[r]);
    }
}

// ---------------------------------------------------------------------------
// Host-side orchestration
// ---------------------------------------------------------------------------
extern "C" void kernel_launch(void* const* d_in, const int* in_sizes, int n_in,
                              void* d_out, int out_size, void* d_ws,
                              size_t ws_size, hipStream_t stream) {
  (void)in_sizes; (void)n_in; (void)out_size; (void)ws_size;
  const float* x     = (const float*)d_in[0];
  const float* w_cq  = (const float*)d_in[1];
  const float* g_q   = (const float*)d_in[2];
  const float* w_ckv = (const float*)d_in[3];
  const float* g_kv  = (const float*)d_in[4];
  const float* w_dqn = (const float*)d_in[5];
  const float* w_dqr = (const float*)d_in[6];
  const float* w_dkn = (const float*)d_in[7];
  const float* w_dv  = (const float*)d_in[8];
  const float* w_kr  = (const float*)d_in[9];
  const float* w_out = (const float*)d_in[10];

  char* ws = (char*)d_ws;
  size_t off = 0;
  auto alloc = [&](size_t bytes) -> void* {
    void* p = ws + off;
    off = (off + bytes + 255) & ~(size_t)255;
    return p;
  };

  // bf16 copies (weights stored TRANSPOSED: Bt is N x K)
  __bf16* x_bf    = (__bf16*)alloc((size_t)BT_ * D_ * 2);
  __bf16* wcq_t   = (__bf16*)alloc((size_t)QR_ * D_ * 2);            // (QR,D)
  __bf16* wckv_t  = (__bf16*)alloc((size_t)KVR_ * D_ * 2);           // (KVR,D)
  __bf16* wdqn_t  = (__bf16*)alloc((size_t)(H_ * ND_) * QR_ * 2);    // (2048,QR)
  __bf16* wdqr_t  = (__bf16*)alloc((size_t)(H_ * RD_) * QR_ * 2);    // (1024,QR)
  __bf16* wdkn_t  = (__bf16*)alloc((size_t)(H_ * ND_) * KVR_ * 2);   // (2048,KVR)
  __bf16* wdv_t   = (__bf16*)alloc((size_t)(H_ * VD_) * KVR_ * 2);   // (2048,KVR)
  __bf16* wkr_t   = (__bf16*)alloc((size_t)RD_ * D_ * 2);            // (64,D)
  __bf16* wout_t  = (__bf16*)alloc((size_t)D_ * (H_ * VD_) * 2);     // (D,2048)
  // f32 intermediates
  float* qlat_raw  = (float*)alloc((size_t)BT_ * QR_ * 4);
  float* kvlat_raw = (float*)alloc((size_t)BT_ * KVR_ * 4);
  float* kr_raw    = (float*)alloc((size_t)BT_ * RD_ * 4);
  float* kr_rope   = (float*)alloc((size_t)BT_ * RD_ * 4);
  float* q_nope    = (float*)alloc((size_t)BT_ * H_ * ND_ * 4);
  float* q_rraw    = (float*)alloc((size_t)BT_ * H_ * RD_ * 4);
  float* k_nope    = (float*)alloc((size_t)BT_ * H_ * ND_ * 4);
  float* v_raw     = (float*)alloc((size_t)BT_ * H_ * VD_ * 4);
  // bf16 activations
  __bf16* qlat_bf  = (__bf16*)alloc((size_t)BT_ * QR_ * 2);
  __bf16* kvlat_bf = (__bf16*)alloc((size_t)BT_ * KVR_ * 2);
  __bf16* q_bf     = (__bf16*)alloc((size_t)B_ * H_ * T_ * DQK_ * 2);
  __bf16* k_bf     = (__bf16*)alloc((size_t)B_ * H_ * T_ * DQK_ * 2);
  __bf16* v_bt     = (__bf16*)alloc((size_t)B_ * H_ * VD_ * T_ * 2);
  __bf16* attn_bf  = (__bf16*)alloc((size_t)BT_ * H_ * VD_ * 2);

  // d_out partitions (all f32): out | k | v_t
  float* out_p = (float*)d_out;
  float* k_p   = out_p + (size_t)BT_ * D_;
  float* v_p   = k_p + (size_t)B_ * H_ * T_ * DQK_;

  auto cvtT = [&](const float* src, __bf16* dst, int R, int Ccol) {
    dim3 g(Ccol / 32, R / 32);
    cvt_transpose_kernel<<<g, 256, 0, stream>>>(src, dst, R, Ccol);
  };
  auto gemm = [&](const __bf16* A, const __bf16* Bt, float* C, int M, int N,
                  int K) {
    dim3 g((N + 127) / 128, M / 128);
    gemm_bf16_kernel<<<g, 256, 0, stream>>>(A, Bt, C, M, N, K);
  };

  // --- stage 0: precision conversion (+ weight transpose) ---
  cvt_f32_bf16_kernel<<<4096, 256, 0, stream>>>(x, x_bf, (size_t)BT_ * D_);
  cvtT(w_cq,  wcq_t,  D_,  QR_);
  cvtT(w_ckv, wckv_t, D_,  KVR_);
  cvtT(w_dqn, wdqn_t, QR_, H_ * ND_);
  cvtT(w_dqr, wdqr_t, QR_, H_ * RD_);
  cvtT(w_dkn, wdkn_t, KVR_, H_ * ND_);
  cvtT(w_dv,  wdv_t,  KVR_, H_ * VD_);
  cvtT(w_kr,  wkr_t,  D_,  RD_);
  cvtT(w_out, wout_t, H_ * VD_, D_);

  // --- stage 1: latent projections + k_rope ---
  gemm(x_bf, wcq_t,  qlat_raw,  BT_, QR_,  D_);
  gemm(x_bf, wckv_t, kvlat_raw, BT_, KVR_, D_);
  gemm(x_bf, wkr_t,  kr_raw,    BT_, RD_,  D_);

  // --- stage 2: rmsnorm -> bf16 latents; rope k_rope ---
  rmsnorm_kernel<<<BT_, 256, 0, stream>>>(qlat_raw, g_q, qlat_bf, QR_);
  rmsnorm_kernel<<<BT_, 256, 0, stream>>>(kvlat_raw, g_kv, kvlat_bf, KVR_);
  rope_k_kernel<<<BT_, 32, 0, stream>>>(kr_raw, kr_rope);

  // --- stage 3: decode projections ---
  gemm(qlat_bf,  wdqn_t, q_nope, BT_, H_ * ND_, QR_);
  gemm(qlat_bf,  wdqr_t, q_rraw, BT_, H_ * RD_, QR_);
  gemm(kvlat_bf, wdkn_t, k_nope, BT_, H_ * ND_, KVR_);
  gemm(kvlat_bf, wdv_t,  v_raw,  BT_, H_ * VD_, KVR_);

  // --- stage 4: rope q + pack q/k/v (also writes f32 k and v_t outputs) ---
  rope_pack_q_kernel<<<B_ * H_ * T_, DQK_, 0, stream>>>(q_nope, q_rraw, q_bf);
  pack_kv_kernel<<<B_ * H_ * T_, DQK_, 0, stream>>>(k_nope, kr_rope, v_raw,
                                                    k_p, v_p, k_bf, v_bt);

  // --- stage 5: flash attention ---
  flash_attn_kernel<<<dim3(T_ / 128, H_, B_), 256, 0, stream>>>(q_bf, k_bf,
                                                                v_bt, attn_bf);

  // --- stage 6: output projection -> d_out ---
  gemm(attn_bf, wout_t, out_p, BT_, D_, H_ * VD_);
}
